// PSABlock_EDFFN_17248588661071
// MI455X (gfx1250) — matrix operand order repaired and checked by reading, compiled, not run on hardware
//
#include <hip/hip_runtime.h>
#include <hip/hip_bf16.h>
#include <math.h>

// ---------------------------------------------------------------------------
// PSABlock + EDFFN for MI455X (gfx1250).  bf16 WMMA (f32 accumulate) for all
// GEMM-shaped work, 16B-granular LDS staging (async-to-LDS when available),
// wide b128 fragment loads with one contiguous-k order for A and B.
// ---------------------------------------------------------------------------

typedef __attribute__((ext_vector_type(16))) __bf16 v16bf;
typedef __attribute__((ext_vector_type(8)))  __bf16 v8bf;
typedef __attribute__((ext_vector_type(8)))  float  v8f;
typedef __attribute__((__vector_size__(4 * sizeof(int)))) int v4i_vs;

#define B_    8
#define C_    256
#define H_    32
#define W_    32
#define N_    1024
#define NH_   4
#define KD_   32
#define HD_   64
#define HID_  512
#define PI_M  1024
#define SCALE_ 0.17677669529663687f   // 32^-0.5
#define TLDA  40                      // padded LDS stride: 80B rows, 16B aligned

// ---- optional gfx1250 async global->LDS path --------------------------------
#if __has_builtin(__builtin_amdgcn_global_load_async_to_lds_b128)
#define ASYNC_LDS 1
static __device__ __forceinline__ void async_cp16(const __bf16* g, __bf16* l) {
  __builtin_amdgcn_global_load_async_to_lds_b128((v4i_vs*)g, (v4i_vs*)l, 0, 0);
}
static __device__ __forceinline__ void async_wait0() {
#if __has_builtin(__builtin_amdgcn_s_wait_asynccnt)
  __builtin_amdgcn_s_wait_asynccnt(0);
#else
  asm volatile("s_wait_asynccnt 0" ::: "memory");
#endif
}
#endif

// ---- WMMA fragment builders (contiguous-k order, k = 16*half + s, A and B alike) ----
// 16-bit A (16x32): lane L -> row=L%16, half=L/16; per lane the 16 halves are
// K = half*16 .. half*16+15, contiguous.
static __device__ __forceinline__ v16bf fragA(const __bf16* base, int lda) {
  int lane = threadIdx.x & 31;
  int half = lane >> 4, row = lane & 15;
  const __bf16* p = base + row * lda + half * 16;
  v8bf lo = *(const v8bf*)(p);        // K = half*16 .. +7
  v8bf hi = *(const v8bf*)(p + 8);    // K = half*16 + 8 .. +15
  return __builtin_shufflevector(lo, hi, 0, 1, 2, 3, 4, 5, 6, 7,
                                 8, 9, 10, 11, 12, 13, 14, 15);
}

// 16-bit B (32x16) from n-major (transposed) storage: lane L -> n=L%16,
// half=L/16; the 16 halves are K = half*16 .. half*16+15, contiguous.
static __device__ __forceinline__ v16bf fragB_nm(const __bf16* base, int ldn) {
  int lane = threadIdx.x & 31;
  int half = lane >> 4, n = lane & 15;
  const __bf16* p = base + n * ldn + half * 16;
  v8bf lo = *(const v8bf*)(p);
  v8bf hi = *(const v8bf*)(p + 8);
  return __builtin_shufflevector(lo, hi, 0, 1, 2, 3, 4, 5, 6, 7,
                                 8, 9, 10, 11, 12, 13, 14, 15);
}

// ---------------------------------------------------------------------------
__global__ void cvt_bf16_kernel(const float* __restrict__ in,
                                __bf16* __restrict__ out, int n) {
  int i = blockIdx.x * blockDim.x + threadIdx.x;
  if (i < n) out[i] = (__bf16)in[i];
}

// RMS-style LN over channel dim; emits bf16 activations [B][C][N]
__global__ void ln_kernel(const float* __restrict__ x, const float* __restrict__ w,
                          const float* __restrict__ bias, __bf16* __restrict__ out) {
  int t = blockIdx.x * blockDim.x + threadIdx.x;
  if (t >= B_ * N_) return;
  int b = t >> 10, n = t & 1023;
  const float* xb = x + (size_t)b * C_ * N_ + n;
  float ss = 0.f;
  for (int c = 0; c < C_; ++c) { float u = xb[(size_t)c * N_]; ss += u * u; }
  float r = rsqrtf(ss * (1.0f / C_) + 1e-6f);
  __bf16* ob = out + (size_t)b * C_ * N_ + n;
  for (int c = 0; c < C_; ++c)
    ob[(size_t)c * N_] = (__bf16)(xb[(size_t)c * N_] * r * w[c] + bias[c]);
}

// ---------------------------------------------------------------------------
// 64x64-tile bf16 WMMA GEMM, D[m,n] = sum_k A[m,k] * X[b,k,n], fused epilogues.
// mode 0: qkv -> BN, split-scatter q,k (query/key-major) and v (bf16)
// mode 1: proj -> BN then x1 = resid + gamma*val (f32)
// mode 2: plain f32 out [B][M][N]
__global__ __launch_bounds__(128) void gemm_kernel(
    const __bf16* __restrict__ A, const __bf16* __restrict__ X,
    int M, int K, int mode,
    const float* __restrict__ g,  const float* __restrict__ bb,
    const float* __restrict__ mm, const float* __restrict__ vv,
    const float* __restrict__ gamma, const float* __restrict__ resid,
    float* __restrict__ fout,
    __bf16* __restrict__ qout, __bf16* __restrict__ kout,
    __bf16* __restrict__ vout) {
  __shared__ __attribute__((aligned(16))) __bf16 ldsA[64 * TLDA];
  __shared__ __attribute__((aligned(16))) __bf16 ldsB[64 * TLDA];
  int b    = blockIdx.z;
  int mblk = blockIdx.y * 64;
  int nblk = blockIdx.x * 64;
  int t    = threadIdx.x;
  int wave = t >> 5;
  int lane = t & 31;
  const __bf16* Xb = X + (size_t)b * K * N_;

  int rowA = t >> 1, khA = (t & 1) * 16;          // A: 32B per thread
  int kkB  = t >> 2, nnB = (t & 3) * 16;          // B: 32B per thread

  v8f acc[4] = {};
  for (int k0 = 0; k0 < K; k0 += 32) {
    const __bf16* gA = A + (size_t)(mblk + rowA) * K + k0 + khA;
    const __bf16* gB = Xb + (size_t)(k0 + kkB) * N_ + nblk + nnB;
#ifdef ASYNC_LDS
    async_cp16(gA,     &ldsA[rowA * TLDA + khA]);
    async_cp16(gA + 8, &ldsA[rowA * TLDA + khA + 8]);
#else
    *(v8bf*)&ldsA[rowA * TLDA + khA]     = *(const v8bf*)(gA);
    *(v8bf*)&ldsA[rowA * TLDA + khA + 8] = *(const v8bf*)(gA + 8);
#endif
    // B tile: wide loads, register-staged transpose into n-major LDS
    v8bf b0 = *(const v8bf*)(gB);
    v8bf b1 = *(const v8bf*)(gB + 8);
#pragma unroll
    for (int i = 0; i < 8; ++i) {
      ldsB[(nnB + i) * TLDA + kkB]     = b0[i];
      ldsB[(nnB + 8 + i) * TLDA + kkB] = b1[i];
    }
    if (k0 + 32 < K) {                 // prefetch next K-tile
      __builtin_prefetch(gA + 32, 0, 3);
      __builtin_prefetch(Xb + (size_t)(k0 + 32 + kkB) * N_ + nblk + nnB, 0, 3);
    }
#ifdef ASYNC_LDS
    async_wait0();
#endif
    __syncthreads();
    v16bf af = fragA(&ldsA[(wave * 16) * TLDA], TLDA);
#pragma unroll
    for (int tt = 0; tt < 4; ++tt) {
      v16bf bf = fragB_nm(&ldsB[(tt * 16) * TLDA], TLDA);
      acc[tt] = __builtin_amdgcn_wmma_f32_16x16x32_bf16(
          false, af, false, bf, (short)0, acc[tt], false, false);
    }
    __syncthreads();
  }

  int half = lane >> 4, col = lane & 15;
#pragma unroll
  for (int tt = 0; tt < 4; ++tt) {
    int n = nblk + tt * 16 + col;
#pragma unroll
    for (int r = 0; r < 8; ++r) {
      int m = mblk + wave * 16 + r + 8 * half;
      float val = acc[tt][r];
      if (mode == 0) {            // qkv + BN + head split
        float s = g[m] * rsqrtf(vv[m] + 1e-5f);
        val = val * s + (bb[m] - mm[m] * s);
        int h = m >> 7, rr = m & 127;
        if (rr < 32)
          qout[(((size_t)(b * NH_ + h)) * N_ + n) * KD_ + rr] = (__bf16)val;
        else if (rr < 64)
          kout[(((size_t)(b * NH_ + h)) * N_ + n) * KD_ + (rr - 32)] = (__bf16)val;
        else
          vout[((size_t)b * C_ + h * HD_ + (rr - 64)) * N_ + n] = (__bf16)val;
      } else if (mode == 1) {     // proj + BN + gamma1 residual
        float s = g[m] * rsqrtf(vv[m] + 1e-5f);
        val = val * s + (bb[m] - mm[m] * s);
        size_t idx = ((size_t)b * C_ + m) * N_ + n;
        fout[idx] = resid[idx] + gamma[m] * val;
      } else {                    // plain f32
        fout[((size_t)b * M + m) * N_ + n] = val;
      }
    }
  }
}

// ---------------------------------------------------------------------------
// Fused attention: one wave handles 16 query rows x all 1024 keys for one
// (b,h).  Scores via WMMA into a 16x1024 f32 LDS slab, exact softmax, then
// AV via WMMA over the 1024-key dimension.  q,k stored [b][h][n][kd] so both
// operands are 2x b128 loads per lane.
__global__ __launch_bounds__(32) void attn_kernel(
    const __bf16* __restrict__ q, const __bf16* __restrict__ kmat,
    const __bf16* __restrict__ vmat, float* __restrict__ o) {
  __shared__ float P[16 * N_];   // 64 KB of the 320 KB WGP LDS
  int n0 = blockIdx.x * 16;
  int h  = blockIdx.y;
  int b  = blockIdx.z;
  int lane = threadIdx.x;
  int half = lane >> 4, col = lane & 15;

  const __bf16* qb = q    + (((size_t)(b * NH_ + h)) * N_ + n0) * KD_;  // [n][kd]
  const __bf16* kb = kmat + ((size_t)(b * NH_ + h)) * N_ * KD_;         // [m][kd]
  const __bf16* vb = vmat + ((size_t)b * C_ + h * HD_) * N_;            // [d][m]

  v16bf qa = fragA(qb, KD_);                 // 16 queries x K=32 (full head dim)
  for (int mt = 0; mt < 64; ++mt) {          // key tiles of 16
    v16bf kf = fragB_nm(kb + (size_t)(mt * 16) * KD_, KD_);
    v8f s = {};
    s = __builtin_amdgcn_wmma_f32_16x16x32_bf16(
        false, qa, false, kf, (short)0, s, false, false);
#pragma unroll
    for (int r = 0; r < 8; ++r)
      P[(r + 8 * half) * N_ + mt * 16 + col] = s[r] * SCALE_;
  }
  __syncthreads();

  // exact softmax over the 1024-key axis, lane-parallel per row
  for (int r = 0; r < 16; ++r) {
    float mx = -1e30f;
    for (int i = lane; i < N_; i += 32) mx = fmaxf(mx, P[r * N_ + i]);
#pragma unroll
    for (int off = 16; off > 0; off >>= 1) mx = fmaxf(mx, __shfl_xor(mx, off, 32));
    float sum = 0.f;
    for (int i = lane; i < N_; i += 32) {
      float e = __expf(P[r * N_ + i] - mx);
      P[r * N_ + i] = e;
      sum += e;
    }
#pragma unroll
    for (int off = 16; off > 0; off >>= 1) sum += __shfl_xor(sum, off, 32);
    float inv = 1.0f / sum;
    for (int i = lane; i < N_; i += 32) P[r * N_ + i] *= inv;
  }
  __syncthreads();

  // out[d][n] = sum_m V[d][m] * P[n][m]
  v8f acc[4] = {};
  for (int mt = 0; mt < 32; ++mt) {          // key chunks of 32
    v16bf pb;
#pragma unroll
    for (int r = 0; r < 8; ++r) {
      int k0 = half * 16 + r * 2;
      pb[2 * r]     = (__bf16)P[col * N_ + mt * 32 + k0];
      pb[2 * r + 1] = (__bf16)P[col * N_ + mt * 32 + k0 + 1];
    }
#pragma unroll
    for (int dt = 0; dt < 4; ++dt) {
      v16bf va = fragA(vb + (size_t)(dt * 16) * N_ + mt * 32, N_);
      acc[dt] = __builtin_amdgcn_wmma_f32_16x16x32_bf16(
          false, va, false, pb, (short)0, acc[dt], false, false);
    }
  }
#pragma unroll
  for (int dt = 0; dt < 4; ++dt)
#pragma unroll
    for (int r = 0; r < 8; ++r) {
      int d = dt * 16 + r + 8 * half;
      o[((size_t)b * C_ + h * HD_ + d) * N_ + n0 + col] = acc[dt][r];
    }
}

// ---------------------------------------------------------------------------
// PE: depthwise 3x3 over v image + BN, add attention output, emit bf16
__global__ void pe_add_kernel(const __bf16* __restrict__ vimg,
                              const float* __restrict__ ao,
                              const float* __restrict__ pw,
                              const float* __restrict__ g,  const float* __restrict__ bb,
                              const float* __restrict__ mm, const float* __restrict__ vv,
                              __bf16* __restrict__ out) {
  int t = blockIdx.x * blockDim.x + threadIdx.x;
  if (t >= B_ * C_ * N_) return;
  int n = t & 1023, c = (t >> 10) & 255;
  int hh = n >> 5, ww = n & 31;
  const __bf16* vb = vimg + ((size_t)(t >> 10)) * N_;
  float acc = 0.f;
#pragma unroll
  for (int dh = -1; dh <= 1; ++dh) {
    int y = hh + dh;
    if (y < 0 || y >= H_) continue;
#pragma unroll
    for (int dw = -1; dw <= 1; ++dw) {
      int xw = ww + dw;
      if (xw < 0 || xw >= W_) continue;
      acc += (float)vb[y * W_ + xw] * pw[c * 9 + (dh + 1) * 3 + (dw + 1)];
    }
  }
  float s = g[c] * rsqrtf(vv[c] + 1e-5f);
  out[t] = (__bf16)(ao[t] + acc * s + (bb[c] - mm[c] * s));
}

// depthwise 3x3 over 1024 channels then GELU(x1)*x2 gate -> bf16 [B][512][N]
__global__ void dwgelu_kernel(const float* __restrict__ xin,
                              const float* __restrict__ dw,
                              __bf16* __restrict__ out) {
  int t = blockIdx.x * blockDim.x + threadIdx.x;
  if (t >= B_ * HID_ * N_) return;
  int n = t & 1023, c = (t >> 10) & 511, b = t >> 19;
  int hh = n >> 5, ww = n & 31;
  const float* x0 = xin + ((size_t)b * PI_M + c) * N_;
  const float* x1 = xin + ((size_t)b * PI_M + c + HID_) * N_;
  float a0 = 0.f, a1 = 0.f;
#pragma unroll
  for (int dh = -1; dh <= 1; ++dh) {
    int y = hh + dh;
    if (y < 0 || y >= H_) continue;
#pragma unroll
    for (int dwi = -1; dwi <= 1; ++dwi) {
      int xw = ww + dwi;
      if (xw < 0 || xw >= W_) continue;
      int off = y * W_ + xw, wi = (dh + 1) * 3 + (dwi + 1);
      a0 += x0[off] * dw[c * 9 + wi];
      a1 += x1[off] * dw[(c + HID_) * 9 + wi];
    }
  }
  float ge = 0.5f * a0 * (1.0f + erff(a0 * 0.70710678118654752f));
  out[t] = (__bf16)(ge * a1);
}

// ---------------------------------------------------------------------------
// Per-8x8-tile rFFT2 * real filter, iFFT2, fused gamma2 residual.
__global__ void fft_kernel(const float* __restrict__ f, const float* __restrict__ P,
                           const float* __restrict__ x1, const float* __restrict__ gamma2,
                           float* __restrict__ out) {
  int t = blockIdx.x * blockDim.x + threadIdx.x;
  if (t >= B_ * C_ * 16) return;
  int tile = t & 15, c = (t >> 4) & 255, b = t >> 12;
  int th = tile >> 2, tw = tile & 3;
  const float* fb = f + ((size_t)b * C_ + c) * N_;

  float X[8][8];
  for (int xr = 0; xr < 8; ++xr)
    for (int yc = 0; yc < 8; ++yc)
      X[xr][yc] = fb[(th * 8 + xr) * W_ + tw * 8 + yc];

  const float cr[8] = {1.f, 0.70710678f, 0.f, -0.70710678f,
                       -1.f, -0.70710678f, 0.f, 0.70710678f};
  const float si[8] = {0.f, 0.70710678f, 1.f, 0.70710678f,
                       0.f, -0.70710678f, -1.f, -0.70710678f};

  float Gr[8][5], Gi[8][5];
  for (int u = 0; u < 8; ++u)
    for (int v = 0; v < 5; ++v) {
      float re = 0.f, im = 0.f;
      for (int xr = 0; xr < 8; ++xr)
        for (int yc = 0; yc < 8; ++yc) {
          int ph = (u * xr + v * yc) & 7;
          re += X[xr][yc] * cr[ph];
          im -= X[xr][yc] * si[ph];
        }
      float p = P[c * 40 + u * 5 + v];
      Gr[u][v] = re * p;
      Gi[u][v] = im * p;
    }

  const float* x1b = x1 + ((size_t)b * C_ + c) * N_;
  float* ob = out + ((size_t)b * C_ + c) * N_;
  float gg = gamma2[c];
  for (int xr = 0; xr < 8; ++xr)
    for (int yc = 0; yc < 8; ++yc) {
      float acc = 0.f;
      for (int u = 0; u < 8; ++u)
        for (int vp = 0; vp < 8; ++vp) {
          float gr, gi;
          if (vp <= 4) { gr = Gr[u][vp]; gi = Gi[u][vp]; }
          else { int uu = (8 - u) & 7; gr = Gr[uu][8 - vp]; gi = -Gi[uu][8 - vp]; }
          int ph = (u * xr + vp * yc) & 7;
          acc += gr * cr[ph] - gi * si[ph];
        }
      acc *= (1.0f / 64.0f);
      int idx = (th * 8 + xr) * W_ + tw * 8 + yc;
      ob[idx] = x1b[idx] + gg * acc;
    }
}

// ---------------------------------------------------------------------------
extern "C" void kernel_launch(void* const* d_in, const int* in_sizes, int n_in,
                              void* d_out, int out_size, void* d_ws, size_t ws_size,
                              hipStream_t stream) {
  const float* x      = (const float*)d_in[0];
  const float* n1w    = (const float*)d_in[1];
  const float* n1b    = (const float*)d_in[2];
  const float* n2w    = (const float*)d_in[3];
  const float* n2b    = (const float*)d_in[4];
  const float* gamma1 = (const float*)d_in[5];
  const float* gamma2 = (const float*)d_in[6];
  const float* qkv_w  = (const float*)d_in[7];
  const float* qkv_g  = (const float*)d_in[8];
  const float* qkv_b  = (const float*)d_in[9];
  const float* qkv_m  = (const float*)d_in[10];
  const float* qkv_v  = (const float*)d_in[11];
  const float* pe_w   = (const float*)d_in[12];
  const float* pe_g   = (const float*)d_in[13];
  const float* pe_b   = (const float*)d_in[14];
  const float* pe_m   = (const float*)d_in[15];
  const float* pe_v   = (const float*)d_in[16];
  const float* proj_w = (const float*)d_in[17];
  const float* proj_g = (const float*)d_in[18];
  const float* proj_b = (const float*)d_in[19];
  const float* proj_m = (const float*)d_in[20];
  const float* proj_v = (const float*)d_in[21];
  const float* pi_w   = (const float*)d_in[22];
  const float* dw_w   = (const float*)d_in[23];
  const float* fft_p  = (const float*)d_in[24];
  const float* po_w   = (const float*)d_in[25];

  char* ws = (char*)d_ws;
  size_t off = 0;
  auto alloc = [&](size_t bytes) -> void* {
    void* p = ws + off;
    off += (bytes + 255) & ~(size_t)255;
    return p;
  };
  __bf16* bqkv  = (__bf16*)alloc((size_t)512 * 256 * 2);
  __bf16* bproj = (__bf16*)alloc((size_t)256 * 256 * 2);
  __bf16* bpi   = (__bf16*)alloc((size_t)1024 * 256 * 2);
  __bf16* bpo   = (__bf16*)alloc((size_t)256 * 512 * 2);
  __bf16* xn    = (__bf16*)alloc((size_t)B_ * C_ * N_ * 2);
  __bf16* qb    = (__bf16*)alloc((size_t)B_ * NH_ * N_ * KD_ * 2);
  __bf16* kb    = (__bf16*)alloc((size_t)B_ * NH_ * N_ * KD_ * 2);
  __bf16* vbuf  = (__bf16*)alloc((size_t)B_ * C_ * N_ * 2);
  float*  attn_o= (float*)alloc((size_t)B_ * C_ * N_ * 4);
  __bf16* ao    = (__bf16*)alloc((size_t)B_ * C_ * N_ * 2);
  float*  x1buf = (float*)alloc((size_t)B_ * C_ * N_ * 4);
  __bf16* xn2   = (__bf16*)alloc((size_t)B_ * C_ * N_ * 2);
  float*  ffn1  = (float*)alloc((size_t)B_ * PI_M * N_ * 4);
  __bf16* gated = (__bf16*)alloc((size_t)B_ * HID_ * N_ * 2);
  float*  ffn2  = (float*)alloc((size_t)B_ * C_ * N_ * 4);

  // weight conversion
  cvt_bf16_kernel<<<(512 * 256 + 255) / 256, 256, 0, stream>>>(qkv_w, bqkv, 512 * 256);
  cvt_bf16_kernel<<<(256 * 256 + 255) / 256, 256, 0, stream>>>(proj_w, bproj, 256 * 256);
  cvt_bf16_kernel<<<(1024 * 256 + 255) / 256, 256, 0, stream>>>(pi_w, bpi, 1024 * 256);
  cvt_bf16_kernel<<<(256 * 512 + 255) / 256, 256, 0, stream>>>(po_w, bpo, 256 * 512);

  // LN1
  ln_kernel<<<(B_ * N_ + 255) / 256, 256, 0, stream>>>(x, n1w, n1b, xn);

  // qkv GEMM (M=512, K=256) + BN + head split
  gemm_kernel<<<dim3(16, 8, B_), 128, 0, stream>>>(
      bqkv, xn, 512, 256, 0, qkv_g, qkv_b, qkv_m, qkv_v,
      nullptr, nullptr, nullptr, qb, kb, vbuf);

  // fused attention
  attn_kernel<<<dim3(N_ / 16, NH_, B_), 32, 0, stream>>>(qb, kb, vbuf, attn_o);

  // PE dwconv + BN + add
  pe_add_kernel<<<(B_ * C_ * N_ + 255) / 256, 256, 0, stream>>>(
      vbuf, attn_o, pe_w, pe_g, pe_b, pe_m, pe_v, ao);

  // proj GEMM (M=256, K=256) + BN + gamma1 residual
  gemm_kernel<<<dim3(16, 4, B_), 128, 0, stream>>>(
      bproj, ao, 256, 256, 1, proj_g, proj_b, proj_m, proj_v,
      gamma1, x, x1buf, nullptr, nullptr, nullptr);

  // LN2
  ln_kernel<<<(B_ * N_ + 255) / 256, 256, 0, stream>>>(x1buf, n2w, n2b, xn2);

  // pi GEMM (M=1024, K=256)
  gemm_kernel<<<dim3(16, 16, B_), 128, 0, stream>>>(
      bpi, xn2, 1024, 256, 2, nullptr, nullptr, nullptr, nullptr,
      nullptr, nullptr, ffn1, nullptr, nullptr, nullptr);

  // dwconv + GELU gate
  dwgelu_kernel<<<(B_ * HID_ * N_ + 255) / 256, 256, 0, stream>>>(ffn1, dw_w, gated);

  // po GEMM (M=256, K=512)
  gemm_kernel<<<dim3(16, 4, B_), 128, 0, stream>>>(
      bpo, gated, 256, 512, 2, nullptr, nullptr, nullptr, nullptr,
      nullptr, nullptr, ffn2, nullptr, nullptr, nullptr);

  // per-tile FFT filter + gamma2 residual -> output
  fft_kernel<<<(B_ * C_ * 16 + 63) / 64, 64, 0, stream>>>(
      ffn2, fft_p, x1buf, gamma2, (float*)d_out);
}
